// SentenceEncoder_2499670966439
// MI455X (gfx1250) — matrix operand order repaired and checked by reading, not compiled, not run on hardware
//
#include <hip/hip_runtime.h>
#include <hip/hip_bf16.h>

// ---------------------------------------------------------------------------
// SentenceEncoder for MI455X (gfx1250), wave32 + WMMA bf16.
//   BS=32, SLEN=4096, EM=256.  Dominant GEMM: (131072x256)@(256x256) reduced
//   per-token by tanh()*we_k -> one energy scalar per token.  Bandwidth-bound
//   (~134MB word_embedded, read twice; 2nd read is L2-resident).
// ---------------------------------------------------------------------------

#define BS    32
#define SLEN  4096
#define EM    256
#define NTOK  (BS * SLEN)      // 131072 tokens
#define NTILE (NTOK / 16)      // 8192 16-token tiles
#define LDSTR 264              // bf16 elems per wk row in LDS (132 dwords; %64==4 -> conflict-free, 16B aligned)

typedef __attribute__((ext_vector_type(16))) __bf16 v16bf;
typedef __attribute__((ext_vector_type(8)))  __bf16 v8bf;
typedef __attribute__((ext_vector_type(8)))  float  v8f;

__device__ __forceinline__ float tanh_pos(float x) {  // x >= 0
#if __has_builtin(__builtin_amdgcn_tanhf)
  return __builtin_amdgcn_tanhf(x);                   // native v_tanh_f32
#else
  float e = __builtin_amdgcn_exp2f(x * 2.8853900817779268f);
  return 1.0f - 2.0f * __builtin_amdgcn_rcpf(e + 1.0f);
#endif
}

// ---------------------------------------------------------------------------
// K1: scalar prologue.  q is w_query broadcast -> single scalar
//     c = tanh(relu(wq . q + bq)) . we_q + be
// ---------------------------------------------------------------------------
__global__ void se_k1_scalar(const float* __restrict__ w_query,
                             const float* __restrict__ wq,
                             const float* __restrict__ bq,
                             const float* __restrict__ we,
                             const float* __restrict__ be,
                             float* __restrict__ c_ws) {
  __shared__ float red[256];
  int e = threadIdx.x;
  float dot = 0.f;
  const float* row = wq + e * EM;
  for (int d = 0; d < EM; ++d) dot += w_query[d] * row[d];
  float qp = fmaxf(dot + bq[e], 0.f);
  red[e] = tanh_pos(qp) * we[e];             // we_q = we[0:256]
  __syncthreads();
  for (int s = 128; s > 0; s >>= 1) {
    if (e < s) red[e] += red[e + s];
    __syncthreads();
  }
  if (e == 0) c_ws[0] = red[0] + be[0];
}

// ---------------------------------------------------------------------------
// K2: WMMA energy kernel.  Per 16-token tile (one wave):
//   D = A(16x256 tokens, bf16) x B(256x256 = wk^T, bf16 in LDS), f32 acc,
//   energy[tok] = sum_n tanh(relu(D + bk[n])) * we_k[n], masked, + c.
// nt loop is kept ROLLED so only one N-tile of B is live at a time (the fully
// unrolled version hoisted all 128 B loads and spilled them to scratch).
// amdgpu_waves_per_eu(1): occupancy is LDS-capped (2 blocks/WGP), so give the
// allocator the full VGPR budget and avoid spills.
// ---------------------------------------------------------------------------
__global__ void __launch_bounds__(256)
__attribute__((amdgpu_waves_per_eu(1)))
se_k2_energy(const float* __restrict__ wemb,
             const unsigned char* __restrict__ mask,
             const float* __restrict__ wk,
             const float* __restrict__ bk,
             const float* __restrict__ we,
             const float* __restrict__ c_ws,
             float* __restrict__ energy_out) {
  __shared__ __bf16 lwk[EM * LDSTR];         // 135168 B of the 320KB WGP LDS

  // --- cooperative fill: wk (f32, row-major [n][k]) -> bf16 LDS, padded rows
  for (int idx = threadIdx.x; idx < EM * EM; idx += blockDim.x) {
    int n = idx >> 8, k = idx & 255;
    lwk[n * LDSTR + k] = (__bf16)wk[idx];
  }
  __syncthreads();

  const int lane = threadIdx.x & 31;
  const int wave = threadIdx.x >> 5;
  const int hi   = lane >> 4;                // 0: lanes 0-15, 1: lanes 16-31
  const int lm   = lane & 15;
  const float cadd = c_ws[0];
  const float* wek = we + EM;                // we_k = we[256:512]

  for (int tile = blockIdx.x * 8 + wave; tile < NTILE; tile += gridDim.x * 8) {
    // ---- load A: 16 tokens x 256 dims, packed bf16 per ISA A-layout.
    // lane<16 : row M=lm, K in {kt*32+0..7, kt*32+16..23}
    // lane>=16: row M=lm, K in {kt*32+8..15, kt*32+24..31}
    const float* arow = wemb + (size_t)(tile * 16 + lm) * EM;
    v16bf areg[8];
#pragma unroll
    for (int kt = 0; kt < 8; ++kt) {
      int c0 = kt * 32 + hi * 8;
      float4 f0 = *(const float4*)(arow + c0);
      float4 f1 = *(const float4*)(arow + c0 + 4);
      float4 f2 = *(const float4*)(arow + c0 + 16);
      float4 f3 = *(const float4*)(arow + c0 + 20);
      v16bf a;
      a[0]=(__bf16)f0.x; a[1]=(__bf16)f0.y; a[2]=(__bf16)f0.z; a[3]=(__bf16)f0.w;
      a[4]=(__bf16)f1.x; a[5]=(__bf16)f1.y; a[6]=(__bf16)f1.z; a[7]=(__bf16)f1.w;
      a[8]=(__bf16)f2.x; a[9]=(__bf16)f2.y; a[10]=(__bf16)f2.z; a[11]=(__bf16)f2.w;
      a[12]=(__bf16)f3.x; a[13]=(__bf16)f3.y; a[14]=(__bf16)f3.z; a[15]=(__bf16)f3.w;
      areg[kt] = a;
    }

    float eacc[8];
#pragma unroll
    for (int j = 0; j < 8; ++j) eacc[j] = 0.f;

    // ---- 16 N-tiles (rolled) x 8 K-steps (unrolled) of v_wmma_f32_16x16x32_bf16
#pragma clang loop unroll(disable)
    for (int nt = 0; nt < 16; ++nt) {
      int n = nt * 16 + lm;
      const __bf16* brow = &lwk[n * LDSTR];
      v8f d = {};
#pragma unroll
      for (int kt = 0; kt < 8; ++kt) {
        // B layout (32x16 bf16): lanes 0-15 hold K={0..7,16..23}, lanes 16-31 K={8..15,24..31},
        // column N = lane&15.  B[k][n] = wk[n][k] -> contiguous in LDS row n.
        int kb = kt * 32 + hi * 8;
        v8bf x0 = *(const v8bf*)(brow + kb);
        v8bf x1 = *(const v8bf*)(brow + kb + 16);
        v16bf b = __builtin_shufflevector(x0, x1, 0,1,2,3,4,5,6,7,
                                                  8,9,10,11,12,13,14,15);
        d = __builtin_amdgcn_wmma_f32_16x16x32_bf16(
                false, areg[kt], false, b, (short)0, d, false, false);
      }
      // ---- epilogue on D tile: relu -> tanh -> * we_k[n], accumulate
      float bkn = bk[n];
      float wkn = wek[n];
#pragma unroll
      for (int j = 0; j < 8; ++j) {
        float v = fmaxf(d[j] + bkn, 0.f);
        eacc[j] += tanh_pos(v) * wkn;
      }
    }

    // ---- reduce over the 16 lanes of each half (n-columns of this lane group)
#pragma unroll
    for (int j = 0; j < 8; ++j) {
      float v = eacc[j];
      v += __shfl_xor(v, 1);
      v += __shfl_xor(v, 2);
      v += __shfl_xor(v, 4);
      v += __shfl_xor(v, 8);
      eacc[j] = v;
    }

    // D row M = j + 8*hi -> token tile*16 + M.  Lanes (j) and (j+16) write.
    int t0 = tile * 16;
#pragma unroll
    for (int j = 0; j < 8; ++j) {
      if (lm == j) {
        int tok = t0 + j + 8 * hi;
        float e = mask[tok] ? (eacc[j] + cadd) : 1e-45f;
        energy_out[tok] = e;
      }
    }
  }
}

// ---------------------------------------------------------------------------
// K3a: per-batch softmax stats (max, sum of exp) + zero the output row.
// ---------------------------------------------------------------------------
__global__ void se_k3a_stats(const float* __restrict__ energy,
                             float* __restrict__ red_ws,
                             float* __restrict__ out) {
  __shared__ float red[256];
  int b = blockIdx.x, t = threadIdx.x;
  const float* e = energy + b * SLEN;

  float m = -3.0e38f;
  for (int i = t; i < SLEN; i += 256) m = fmaxf(m, e[i]);
  red[t] = m; __syncthreads();
  for (int s = 128; s > 0; s >>= 1) { if (t < s) red[t] = fmaxf(red[t], red[t+s]); __syncthreads(); }
  float mall = red[0]; __syncthreads();

  float sum = 0.f;
  for (int i = t; i < SLEN; i += 256) sum += __expf(e[i] - mall);
  red[t] = sum; __syncthreads();
  for (int s = 128; s > 0; s >>= 1) { if (t < s) red[t] += red[t+s]; __syncthreads(); }

  if (t == 0) { red_ws[b*2] = mall; red_ws[b*2+1] = red[0]; }
  out[b * EM + t] = 0.f;                     // zero every launch (atomics follow)
}

// ---------------------------------------------------------------------------
// K3b: weighted sum  out[b,e] += sum_s softmax(energy)[b,s] * wemb[b,s,e]
//   grid = 32 batches x 8 s-chunks (word_embedded comes back out of L2).
// ---------------------------------------------------------------------------
__global__ void se_k3b_wsum(const float* __restrict__ wemb,
                            const float* __restrict__ energy,
                            const float* __restrict__ red_ws,
                            float* __restrict__ out) {
  __shared__ float lw[512];
  int b = blockIdx.x >> 3;
  int s0 = (blockIdx.x & 7) * 512;
  int t = threadIdx.x;

  float mall = red_ws[b*2];
  float inv  = 1.0f / red_ws[b*2+1];
  for (int i = t; i < 512; i += 256)
    lw[i] = __expf(energy[b * SLEN + s0 + i] - mall) * inv;
  __syncthreads();

  const float* wp = wemb + ((size_t)b * SLEN + s0) * EM + t;
  float acc = 0.f;
  for (int s = 0; s < 512; ++s) acc += lw[s] * wp[(size_t)s * EM];
  atomicAdd(&out[b * EM + t], acc);
}

// ---------------------------------------------------------------------------
extern "C" void kernel_launch(void* const* d_in, const int* in_sizes, int n_in,
                              void* d_out, int out_size, void* d_ws, size_t ws_size,
                              hipStream_t stream) {
  const float*         wemb    = (const float*)d_in[0];        // (32,4096,256)
  const unsigned char* mask    = (const unsigned char*)d_in[1];// (32,4096) bool
  const float*         w_query = (const float*)d_in[2];        // (256,1)
  const float*         wq      = (const float*)d_in[3];        // (256,256)
  const float*         bq      = (const float*)d_in[4];        // (256,)
  const float*         wk      = (const float*)d_in[5];        // (256,256)
  const float*         bk      = (const float*)d_in[6];        // (256,)
  const float*         we      = (const float*)d_in[7];        // (1,512)
  const float*         be      = (const float*)d_in[8];        // (1,)
  float* out = (float*)d_out;                                  // (32,256) f32

  float* c_ws   = (float*)d_ws;                                // 1 float
  float* energy = (float*)((char*)d_ws + 64);                  // 131072 floats
  float* red_ws = (float*)((char*)d_ws + 64 + NTOK * sizeof(float)); // 64 floats

  se_k1_scalar<<<1, 256, 0, stream>>>(w_query, wq, bq, we, be, c_ws);
  se_k2_energy<<<256, 256, 0, stream>>>(wemb, mask, wk, bk, we, c_ws, energy);
  se_k3a_stats<<<BS, 256, 0, stream>>>(energy, red_ws, out);
  se_k3b_wsum<<<BS * 8, 256, 0, stream>>>(wemb, energy, red_ws, out);
}
